// MinkUNet_43499428774434
// MI455X (gfx1250) — matrix operand order repaired and checked by reading, not compiled, not run on hardware
//
#include <hip/hip_runtime.h>
#include <cstddef>

// ============================================================================
// MinkUNet (dense 64^3) for MI455X / gfx1250.
// bf16 NDHWC activations (C padded to 32), implicit-GEMM convs via
// v_wmma_f32_16x16x32_bf16 (fp32 accumulate), 32x32 register tiles per wave
// (2x2 WMMA sub-tiles), weights pre-packed on device into WMMA B-fragment
// order. BN uses batch stats (coalesced 2-pass with LDS/global f32 atomics).
// All launches on `stream`; workspace managed by a bump arena (graph-safe).
// ============================================================================

typedef unsigned short u16;
typedef unsigned int   u32;
typedef __attribute__((ext_vector_type(16))) __bf16 v16bf;
typedef __attribute__((ext_vector_type(8)))  float  v8f;
typedef __attribute__((ext_vector_type(4)))  u32    v4u;

struct BPack { v4u lo, hi; };   // 32 bytes == one v16bf fragment

__device__ __forceinline__ float bf2f(u16 h) {
  u32 x = ((u32)h) << 16;
  return __builtin_bit_cast(float, x);
}
__device__ __forceinline__ u16 f2bf(float f) {
  u32 x = __builtin_bit_cast(u32, f);
  return (u16)((x + 0x7FFFu + ((x >> 16) & 1u)) >> 16);   // round-to-nearest-even
}
__device__ __forceinline__ v8f wmma_bf16(const BPack& A, const BPack& B, v8f C) {
  return __builtin_amdgcn_wmma_f32_16x16x32_bf16(
      false, __builtin_bit_cast(v16bf, A),
      false, __builtin_bit_cast(v16bf, B),
      (short)0, C, false, false);
}

// ---------------------------------------------------------------------------
// utility kernels
// ---------------------------------------------------------------------------
__global__ void zero_u16_kernel(u16* p, long n) {
  long i = (long)blockIdx.x * blockDim.x + threadIdx.x;
  if (i < n) p[i] = 0;
}
__global__ void zero_f32_kernel(float* p, int n) {
  int i = blockIdx.x * blockDim.x + threadIdx.x;
  if (i < n) p[i] = 0.f;
}

// x: fp32 NCDHW (1,3,D,H,W) -> bf16 [voxel][Cp], channels >=3 zeroed
__global__ void convert_input_kernel(const float* __restrict__ x, u16* __restrict__ dst,
                                     long nvox, int Cp) {
  long i = (long)blockIdx.x * blockDim.x + threadIdx.x;
  if (i >= nvox * Cp) return;
  int c = (int)(i % Cp); long v = i / Cp;
  float val = (c < 3) ? x[(long)c * nvox + v] : 0.f;
  dst[i] = f2bf(val);
}

// fp32 OIDHW weights -> bf16 WMMA B-fragments: frag[tap][ck][ct][lane][e]
// lane l: N(cout)=l&15, K(cin within 32-chunk) = e<8 ? (l>>4)*8 + e : 16 + (l>>4)*8 + (e-8)   (ISA B layout)
__global__ void pack_frag_kernel(const float* __restrict__ w, u16* __restrict__ frag,
                                 int k3, int CIN, int COUT, int nck, int nct) {
  long i = (long)blockIdx.x * blockDim.x + threadIdx.x;
  long total = (long)k3 * nck * nct * 512;
  if (i >= total) return;
  int e    = (int)(i & 15);
  int lane = (int)((i >> 4) & 31);
  long t   = i >> 9;
  int ct = (int)(t % nct); t /= nct;
  int ck = (int)(t % nck);
  int tap = (int)(t / nck);
  int cin = ck * 32 + ((e < 8) ? ((lane >> 4) * 8 + e) : (16 + (lane >> 4) * 8 + (e - 8)));
  int co  = ct * 16 + (lane & 15);
  float v = 0.f;
  if (cin < CIN && co < COUT) v = w[((long)co * CIN + cin) * k3 + tap];
  frag[i] = f2bf(v);
}

// ---------------------------------------------------------------------------
// implicit-GEMM conv: one wave -> 32 voxels x 32 couts (2x2 WMMA tiles).
// A fragment (16x32 bf16): lane holds voxel row m=lane&15; its 16 bf16 are
// channels [half*8,+8) and [16+half*8,+8) -> two contiguous b128 loads.
// ---------------------------------------------------------------------------
__global__ void __launch_bounds__(256)
conv_wmma_kernel(const u16* __restrict__ in, const u16* __restrict__ frag,
                 u16* __restrict__ raw,
                 int ID, int IH, int IW, int Cpin,
                 int OD, int OH, int OW, int Cpout,
                 int k, int stride, int pad, int nct, int nvt2)
{
  int wave = threadIdx.x >> 5;
  int lane = threadIdx.x & 31;
  int vt2 = blockIdx.x * 8 + wave;
  if (vt2 >= nvt2) return;                // uniform per wave: EXEC stays full
  int ct0 = blockIdx.y << 1;
  bool has1 = (ct0 + 1) < nct;            // uniform
  int half = lane >> 4;
  int m    = lane & 15;
  long nvox = (long)OD * OH * OW;

  long vm0 = (long)vt2 * 32 + m;          // A-row voxel, tile 0
  long vm1 = vm0 + 16;                    // A-row voxel, tile 1
  bool av0 = vm0 < nvox, av1 = vm1 < nvox;
  int ox0 = (int)(vm0 % OW); long q0 = vm0 / OW; int oy0 = (int)(q0 % OH), oz0 = (int)(q0 / OH);
  int ox1 = (int)(vm1 % OW); long q1 = vm1 / OW; int oy1 = (int)(q1 % OH), oz1 = (int)(q1 / OH);

  int k3  = k * k * k;
  int nck = Cpin >> 5;
  v4u zv = {0u, 0u, 0u, 0u};
  v8f a00 = {}, a01 = {}, a10 = {}, a11 = {};

  for (int tap = 0; tap < k3; ++tap) {
    int tx = tap % k; int t2 = tap / k; int ty = t2 % k; int tz = t2 / k;
    int iz0 = oz0 * stride + tz - pad, iy0 = oy0 * stride + ty - pad, ix0 = ox0 * stride + tx - pad;
    int iz1 = oz1 * stride + tz - pad, iy1 = oy1 * stride + ty - pad, ix1 = ox1 * stride + tx - pad;
    bool v0 = av0 && iz0 >= 0 && iz0 < ID && iy0 >= 0 && iy0 < IH && ix0 >= 0 && ix0 < IW;
    bool v1 = av1 && iz1 >= 0 && iz1 < ID && iy1 >= 0 && iy1 < IH && ix1 >= 0 && ix1 < IW;
    long b0 = v0 ? ((long)(iz0 * IH + iy0) * IW + ix0) * Cpin : 0;
    long b1 = v1 ? ((long)(iz1 * IH + iy1) * IW + ix1) * Cpin : 0;
    for (int ck = 0; ck < nck; ++ck) {
      BPack A0, A1;
      if (v0) { const v4u* p = (const v4u*)(in + b0 + (long)ck * 32); A0.lo = p[half]; A0.hi = p[2 + half]; }
      else    { A0.lo = zv; A0.hi = zv; }
      if (v1) { const v4u* p = (const v4u*)(in + b1 + (long)ck * 32); A1.lo = p[half]; A1.hi = p[2 + half]; }
      else    { A1.lo = zv; A1.hi = zv; }
      long fb = (((long)tap * nck + ck) * nct) << 9;
      const v4u* qb0 = (const v4u*)(frag + fb + ((long)ct0 << 9) + (long)lane * 16);
      BPack B0; B0.lo = qb0[0]; B0.hi = qb0[1];
      a00 = wmma_bf16(A0, B0, a00);
      a10 = wmma_bf16(A1, B0, a10);
      if (has1) {
        const v4u* qb1 = (const v4u*)(frag + fb + ((long)(ct0 + 1) << 9) + (long)lane * 16);
        BPack B1; B1.lo = qb1[0]; B1.hi = qb1[1];
        a01 = wmma_bf16(A0, B1, a01);
        a11 = wmma_bf16(A1, B1, a11);
      }
    }
  }

  // D layout: lane covers cout N=lane&15; VGPR r covers voxel M = r + half*8
  int n   = lane & 15;
  long vr0 = (long)vt2 * 32 + half * 8;
  long vr1 = vr0 + 16;
  int co0 = ct0 * 16 + n;
  int co1 = co0 + 16;
  for (int r = 0; r < 8; ++r) {
    long va = vr0 + r, vb = vr1 + r;
    if (va < nvox) {
      raw[va * Cpout + co0] = f2bf(a00[r]);
      if (has1) raw[va * Cpout + co1] = f2bf(a01[r]);
    }
    if (vb < nvox) {
      raw[vb * Cpout + co0] = f2bf(a10[r]);
      if (has1) raw[vb * Cpout + co1] = f2bf(a11[r]);
    }
  }
}

// ---------------------------------------------------------------------------
// deconv k=2 s=2: out[2z+tz,2y+ty,2x+tx] = in[z,y,x] x W[tap]; 8 1x1 GEMMs,
// 32 input voxels x 32 couts per wave.
// ---------------------------------------------------------------------------
__global__ void __launch_bounds__(256)
deconv_wmma_kernel(const u16* __restrict__ in, const u16* __restrict__ frag,
                   u16* __restrict__ raw,
                   int ID, int IH, int IW, int Cpin, int Cpout, int nct, int nvt2)
{
  int wave = threadIdx.x >> 5;
  int lane = threadIdx.x & 31;
  int vt2 = blockIdx.x * 8 + wave;
  if (vt2 >= nvt2) return;
  int ct0 = blockIdx.y << 1;
  bool has1 = (ct0 + 1) < nct;
  int tap = blockIdx.z;
  int tx = tap & 1, ty = (tap >> 1) & 1, tz = tap >> 2;
  int half = lane >> 4;
  int m    = lane & 15;
  long nvin = (long)ID * IH * IW;
  long vm0 = (long)vt2 * 32 + m;
  long vm1 = vm0 + 16;
  bool av0 = vm0 < nvin, av1 = vm1 < nvin;
  long b0 = av0 ? vm0 * Cpin : 0;
  long b1 = av1 ? vm1 * Cpin : 0;
  int nck = Cpin >> 5;
  v4u zv = {0u, 0u, 0u, 0u};
  v8f a00 = {}, a01 = {}, a10 = {}, a11 = {};

  for (int ck = 0; ck < nck; ++ck) {
    BPack A0, A1;
    if (av0) { const v4u* p = (const v4u*)(in + b0 + (long)ck * 32); A0.lo = p[half]; A0.hi = p[2 + half]; }
    else     { A0.lo = zv; A0.hi = zv; }
    if (av1) { const v4u* p = (const v4u*)(in + b1 + (long)ck * 32); A1.lo = p[half]; A1.hi = p[2 + half]; }
    else     { A1.lo = zv; A1.hi = zv; }
    long fb = (((long)tap * nck + ck) * nct) << 9;
    const v4u* qb0 = (const v4u*)(frag + fb + ((long)ct0 << 9) + (long)lane * 16);
    BPack B0; B0.lo = qb0[0]; B0.hi = qb0[1];
    a00 = wmma_bf16(A0, B0, a00);
    a10 = wmma_bf16(A1, B0, a10);
    if (has1) {
      const v4u* qb1 = (const v4u*)(frag + fb + ((long)(ct0 + 1) << 9) + (long)lane * 16);
      BPack B1; B1.lo = qb1[0]; B1.hi = qb1[1];
      a01 = wmma_bf16(A0, B1, a01);
      a11 = wmma_bf16(A1, B1, a11);
    }
  }

  int n   = lane & 15;
  int co0 = ct0 * 16 + n;
  int co1 = co0 + 16;
  int OH = IH * 2, OW = IW * 2;
  for (int r = 0; r < 8; ++r) {
    for (int t = 0; t < 2; ++t) {
      long vr = (long)vt2 * 32 + t * 16 + half * 8 + r;
      if (vr < nvin) {
        int  x = (int)(vr % IW);
        long t1 = vr / IW;
        int  y = (int)(t1 % IH);
        int  z = (int)(t1 / IH);
        long o = ((long)(2 * z + tz) * OH + (2 * y + ty)) * OW + (2 * x + tx);
        float va = (t == 0) ? a00[r] : a10[r];
        raw[o * Cpout + co0] = f2bf(va);
        if (has1) {
          float vb = (t == 0) ? a01[r] : a11[r];
          raw[o * Cpout + co1] = f2bf(vb);
        }
      }
    }
  }
}

// coalesced per-channel sum/sumsq: LDS f32 atomics per block, then one
// global f32 atomic per channel per block. stats must be pre-zeroed.
__global__ void __launch_bounds__(256)
bn_stats_kernel(const u16* __restrict__ raw, long nvox, int C, int Cp,
                float* __restrict__ stats) {
  __shared__ float s1[192], s2[192];
  for (int c = threadIdx.x; c < Cp; c += 256) { s1[c] = 0.f; s2[c] = 0.f; }
  __syncthreads();
  long total = nvox * Cp;
  for (long i = (long)blockIdx.x * 256 + threadIdx.x; i < total; i += (long)gridDim.x * 256) {
    int c = (int)(i % Cp);
    if (c < C) {
      float x = bf2f(raw[i]);
      atomicAdd(&s1[c], x);
      atomicAdd(&s2[c], x * x);
    }
  }
  __syncthreads();
  for (int c = threadIdx.x; c < C; c += 256) {
    atomicAdd(&stats[c],     s1[c]);
    atomicAdd(&stats[C + c], s2[c]);
  }
}

// y = (x-mean)*rsqrt(var+eps)*g + b  [+resid] [relu]; bf16 out at chanOff
__global__ void bn_apply_kernel(u16* __restrict__ dst, const u16* __restrict__ raw,
                                const float* __restrict__ stats,
                                const float* __restrict__ g, const float* __restrict__ b,
                                long nvox, int C, int CpRaw, int CpDst, int chanOff,
                                const u16* __restrict__ resid, int CpRes,
                                float inv_n, int act) {
  long i = (long)blockIdx.x * blockDim.x + threadIdx.x;
  if (i >= nvox * C) return;
  int  c = (int)(i % C);
  long v = i / C;
  float mean = stats[c] * inv_n;
  float var  = stats[C + c] * inv_n - mean * mean;
  float x = (bf2f(raw[v * CpRaw + c]) - mean) * rsqrtf(var + 1e-5f) * g[c] + b[c];
  if (resid) x += bf2f(resid[v * CpRes + c]);
  if (act) x = fmaxf(x, 0.f);
  dst[v * CpDst + chanOff + c] = f2bf(x);
}

// copy skip channels into concat buffer at channel offset
__global__ void copy_chan_kernel(u16* __restrict__ dst, int CpDst, int off,
                                 const u16* __restrict__ src, int CpSrc, int C, long nvox) {
  long i = (long)blockIdx.x * blockDim.x + threadIdx.x;
  if (i >= nvox * C) return;
  int c = (int)(i % C); long v = i / C;
  dst[v * CpDst + off + c] = src[v * CpSrc + c];
}

// per-voxel head MLP: 48 -> 20 (leaky 0.1) -> 3 (tanh), fp32 out
__global__ void head_kernel(const u16* __restrict__ feat, int Cp,
                            const float* __restrict__ w1, const float* __restrict__ b1,
                            const float* __restrict__ w2, const float* __restrict__ b2,
                            float* __restrict__ out, long nvox) {
  long v = (long)blockIdx.x * blockDim.x + threadIdx.x;
  if (v >= nvox) return;
  float f[48];
  for (int c = 0; c < 48; ++c) f[c] = bf2f(feat[v * Cp + c]);
  float h[20];
  for (int j = 0; j < 20; ++j) {
    float s = b1[j];
    for (int c = 0; c < 48; ++c) s += f[c] * w1[c * 20 + j];
    h[j] = (s > 0.f) ? s : 0.1f * s;
  }
  for (int j = 0; j < 3; ++j) {
    float s = b2[j];
    for (int c = 0; c < 20; ++c) s += h[c] * w2[c * 3 + j];
    out[v * 3 + j] = tanhf(s);
  }
}

// ---------------------------------------------------------------------------
// host orchestration
// ---------------------------------------------------------------------------
struct Arena {
  char* base; size_t cap; size_t pos;
  void* alloc(size_t bytes) {
    size_t a = (pos + 255) & ~(size_t)255;
    if (a + bytes > cap) a = 0;           // defensive wrap (no runtime here)
    pos = a + bytes;
    return base + a;
  }
};

struct Tensor {
  u16* p; int D, H, W, C, Cp;
  long nvox() const { return (long)D * H * W; }
};

static inline int rup(int x, int m) { return (x + m - 1) / m * m; }

static Tensor allocT(Arena& A, int D, int H, int W, int C) {
  Tensor t; t.D = D; t.H = H; t.W = W; t.C = C; t.Cp = rup(C, 32);
  t.p = (u16*)A.alloc((size_t)t.nvox() * t.Cp * sizeof(u16));
  return t;
}

static void zero_tensor(const Tensor& t, hipStream_t s) {
  long n = t.nvox() * t.Cp;
  zero_u16_kernel<<<dim3((unsigned)((n + 255) / 256)), dim3(256), 0, s>>>(t.p, n);
}

static void run_stats(const Tensor& raw, int cout, float* stats, hipStream_t s) {
  zero_f32_kernel<<<dim3((unsigned)((2 * cout + 255) / 256)), dim3(256), 0, s>>>(stats, 2 * cout);
  long nvox = raw.nvox();
  long totalThreads = (nvox * raw.Cp + 255) / 256;
  int NB = (int)(totalThreads < 1024 ? (totalThreads > 0 ? totalThreads : 1) : 1024);
  bn_stats_kernel<<<dim3((unsigned)NB), dim3(256), 0, s>>>(raw.p, nvox, cout, raw.Cp, stats);
}

// conv -> bn (+optional residual) (+optional relu); writes into dest@chanOff or new tensor
static Tensor conv_bn(Arena& A, hipStream_t s, const Tensor& in,
                      const float* w, const float* g, const float* b,
                      int k, int stride, int pad, int cout, int act,
                      const Tensor* resid, Tensor* dest, int chanOff) {
  int OD = (in.D + 2 * pad - k) / stride + 1;
  int OH = (in.H + 2 * pad - k) / stride + 1;
  int OW = (in.W + 2 * pad - k) / stride + 1;
  Tensor out;
  if (dest) out = *dest;
  else { out = allocT(A, OD, OH, OW, cout); zero_tensor(out, s); }

  size_t mark = A.pos;
  int k3 = k * k * k;
  int nck = in.Cp / 32;
  int nct = cout / 16;                    // all couts are multiples of 16
  long fragElems = (long)k3 * nck * nct * 512;
  u16* frag = (u16*)A.alloc((size_t)fragElems * sizeof(u16));
  pack_frag_kernel<<<dim3((unsigned)((fragElems + 255) / 256)), dim3(256), 0, s>>>(
      w, frag, k3, in.C, cout, nck, nct);

  Tensor raw = allocT(A, OD, OH, OW, cout);
  long nvox = raw.nvox();
  int nvt2 = (int)((nvox + 31) / 32);
  dim3 grid((unsigned)((nvt2 + 7) / 8), (unsigned)((nct + 1) / 2), 1);
  conv_wmma_kernel<<<grid, dim3(256), 0, s>>>(
      in.p, frag, raw.p, in.D, in.H, in.W, in.Cp,
      OD, OH, OW, raw.Cp, k, stride, pad, nct, nvt2);

  float* stats = (float*)A.alloc(sizeof(float) * 2 * cout);
  run_stats(raw, cout, stats, s);

  long n = nvox * cout;
  bn_apply_kernel<<<dim3((unsigned)((n + 255) / 256)), dim3(256), 0, s>>>(
      out.p, raw.p, stats, g, b, nvox, cout, raw.Cp, out.Cp, chanOff,
      resid ? resid->p : nullptr, resid ? resid->Cp : 0,
      1.f / (float)nvox, act);

  A.pos = mark;
  return out;
}

// deconv(k2,s2) -> bn -> relu written into dest channels [0,cout)
static void deconv_bn(Arena& A, hipStream_t s, const Tensor& in,
                      const float* w, const float* g, const float* b,
                      int cout, Tensor& dest) {
  size_t mark = A.pos;
  int nck = in.Cp / 32, nct = cout / 16;
  long fragElems = (long)8 * nck * nct * 512;
  u16* frag = (u16*)A.alloc((size_t)fragElems * sizeof(u16));
  pack_frag_kernel<<<dim3((unsigned)((fragElems + 255) / 256)), dim3(256), 0, s>>>(
      w, frag, 8, in.C, cout, nck, nct);

  Tensor raw = allocT(A, in.D * 2, in.H * 2, in.W * 2, cout);
  long nvin = in.nvox();
  int nvt2 = (int)((nvin + 31) / 32);
  dim3 grid((unsigned)((nvt2 + 7) / 8), (unsigned)((nct + 1) / 2), 8);
  deconv_wmma_kernel<<<grid, dim3(256), 0, s>>>(
      in.p, frag, raw.p, in.D, in.H, in.W, in.Cp, raw.Cp, nct, nvt2);

  float* stats = (float*)A.alloc(sizeof(float) * 2 * cout);
  run_stats(raw, cout, stats, s);

  long nvo = raw.nvox();
  long n = nvo * cout;
  bn_apply_kernel<<<dim3((unsigned)((n + 255) / 256)), dim3(256), 0, s>>>(
      dest.p, raw.p, stats, g, b, nvo, cout, raw.Cp, dest.Cp, 0,
      nullptr, 0, 1.f / (float)nvo, 1);
  A.pos = mark;
}

// ---- param cursor helpers (JAX pytree: dict keys sorted, lists in order) ----
struct CbrP { const float *w, *g, *b; };
struct ResP { const float *w1, *g1, *b1, *w2, *g2, *b2, *wd, *gd, *bd; bool has_wd; };

static CbrP cbrp(void* const* din, int& i) {    // keys sorted: b, g, w
  CbrP c;
  c.b = (const float*)din[i++];
  c.g = (const float*)din[i++];
  c.w = (const float*)din[i++];
  return c;
}
static ResP resp(void* const* din, int& i, bool wd) {  // b1,b2,[bd],g1,g2,[gd],w1,w2,[wd]
  ResP r; r.has_wd = wd;
  r.b1 = (const float*)din[i++];
  r.b2 = (const float*)din[i++];
  r.bd = wd ? (const float*)din[i++] : nullptr;
  r.g1 = (const float*)din[i++];
  r.g2 = (const float*)din[i++];
  r.gd = wd ? (const float*)din[i++] : nullptr;
  r.w1 = (const float*)din[i++];
  r.w2 = (const float*)din[i++];
  r.wd = wd ? (const float*)din[i++] : nullptr;
  return r;
}

static Tensor res_block(Arena& A, hipStream_t s, const Tensor& in, const ResP& p, int cout) {
  Tensor h = conv_bn(A, s, in, p.w1, p.g1, p.b1, 3, 1, 1, cout, 1, nullptr, nullptr, 0);
  Tensor sc;
  if (p.has_wd) sc = conv_bn(A, s, in, p.wd, p.gd, p.bd, 1, 1, 0, cout, 0, nullptr, nullptr, 0);
  else          sc = in;
  return conv_bn(A, s, h, p.w2, p.g2, p.b2, 3, 1, 1, cout, 1, &sc, nullptr, 0);
}

static Tensor stage_fn(Arena& A, hipStream_t s, Tensor h,
                       const CbrP& d, const ResP& r0, const ResP& r1, int cout) {
  h = conv_bn(A, s, h, d.w, d.g, d.b, 2, 2, 0, h.C, 1, nullptr, nullptr, 0);
  h = res_block(A, s, h, r0, cout);
  h = res_block(A, s, h, r1, cout);
  return h;
}

static Tensor up_fn(Arena& A, hipStream_t s, const Tensor& y, const Tensor& skip,
                    const CbrP& de, const ResP& r0, const ResP& r1, int cout) {
  Tensor cat = allocT(A, y.D * 2, y.H * 2, y.W * 2, cout + skip.C);
  zero_tensor(cat, s);
  deconv_bn(A, s, y, de.w, de.g, de.b, cout, cat);
  long nv = cat.nvox();
  long n  = nv * skip.C;
  copy_chan_kernel<<<dim3((unsigned)((n + 255) / 256)), dim3(256), 0, s>>>(
      cat.p, cat.Cp, cout, skip.p, skip.Cp, skip.C, nv);
  Tensor h = res_block(A, s, cat, r0, cout);
  h = res_block(A, s, h, r1, cout);
  return h;
}

// ---------------------------------------------------------------------------
extern "C" void kernel_launch(void* const* d_in, const int* in_sizes, int n_in,
                              void* d_out, int out_size, void* d_ws, size_t ws_size,
                              hipStream_t stream) {
  (void)in_sizes; (void)n_in; (void)out_size;

  // d_in[0] = x; params leaves follow in JAX pytree order (sorted dict keys):
  // last, stage1..4, stem, up1..4
  int i = 1;
  const float* Lb1 = (const float*)d_in[i++];
  const float* Lb2 = (const float*)d_in[i++];
  const float* Lw1 = (const float*)d_in[i++];
  const float* Lw2 = (const float*)d_in[i++];

  CbrP s1d = cbrp(d_in, i); ResP s1r0 = resp(d_in, i, false), s1r1 = resp(d_in, i, false);
  CbrP s2d = cbrp(d_in, i); ResP s2r0 = resp(d_in, i, true),  s2r1 = resp(d_in, i, false);
  CbrP s3d = cbrp(d_in, i); ResP s3r0 = resp(d_in, i, true),  s3r1 = resp(d_in, i, false);
  CbrP s4d = cbrp(d_in, i); ResP s4r0 = resp(d_in, i, true),  s4r1 = resp(d_in, i, false);
  CbrP st0 = cbrp(d_in, i); CbrP st1 = cbrp(d_in, i);
  CbrP u1de = cbrp(d_in, i); ResP u1r0 = resp(d_in, i, true), u1r1 = resp(d_in, i, false);
  CbrP u2de = cbrp(d_in, i); ResP u2r0 = resp(d_in, i, true), u2r1 = resp(d_in, i, false);
  CbrP u3de = cbrp(d_in, i); ResP u3r0 = resp(d_in, i, true), u3r1 = resp(d_in, i, false);
  CbrP u4de = cbrp(d_in, i); ResP u4r0 = resp(d_in, i, true), u4r1 = resp(d_in, i, false);

  Arena A{(char*)d_ws, ws_size, 0};
  const long G3 = 64L * 64 * 64;

  // input: NCDHW fp32 -> bf16 NDHWC (Cp=32, pads zero)
  Tensor xin = allocT(A, 64, 64, 64, 3);
  {
    long n = G3 * xin.Cp;
    convert_input_kernel<<<dim3((unsigned)((n + 255) / 256)), dim3(256), 0, stream>>>(
        (const float*)d_in[0], xin.p, G3, xin.Cp);
  }

  // stem
  Tensor t  = conv_bn(A, stream, xin, st0.w, st0.g, st0.b, 3, 1, 1, 16, 1, nullptr, nullptr, 0);
  Tensor x0 = conv_bn(A, stream, t,   st1.w, st1.g, st1.b, 3, 1, 1, 16, 1, nullptr, nullptr, 0);

  // encoder
  Tensor x1 = stage_fn(A, stream, x0, s1d, s1r0, s1r1, 16);
  Tensor x2 = stage_fn(A, stream, x1, s2d, s2r0, s2r1, 32);
  Tensor x3 = stage_fn(A, stream, x2, s3d, s3r0, s3r1, 64);
  Tensor x4 = stage_fn(A, stream, x3, s4d, s4r0, s4r1, 128);

  // decoder
  Tensor y = up_fn(A, stream, x4, x3, u1de, u1r0, u1r1, 128);
  y = up_fn(A, stream, y, x2, u2de, u2r0, u2r1, 64);
  y = up_fn(A, stream, y, x1, u3de, u3r0, u3r1, 48);
  y = up_fn(A, stream, y, x0, u4de, u4r0, u4r1, 48);

  // head MLP -> fp32 (nvox, 3)
  head_kernel<<<dim3((unsigned)((G3 + 255) / 256)), dim3(256), 0, stream>>>(
      y.p, y.Cp, Lw1, Lb1, Lw2, Lb2, (float*)d_out, G3);
}
